// CrossAttention_146028888568
// MI455X (gfx1250) — matrix operand order repaired and checked by reading, not compiled, not run on hardware
//
#include <hip/hip_runtime.h>

typedef __attribute__((ext_vector_type(16))) _Float16 v16h;
typedef __attribute__((ext_vector_type(8)))  _Float16 v8h;
typedef __attribute__((ext_vector_type(8)))  float    v8f;

#define NB   8
#define SQ   2048
#define SJ   2048
#define QD   512     // query/model dim
#define IDM  256     // inner dim

// ---------------------------------------------------------------------------
// Tensor Data Mover availability (toolchain-dependent arity)
// ---------------------------------------------------------------------------
#if defined(__has_builtin)
#if __has_builtin(__builtin_amdgcn_tensor_load_to_lds)
#define HAVE_TDM 1
#endif
#endif
#ifndef HAVE_TDM
#define HAVE_TDM 0
#endif

#if __has_include(<hip/amd_detail/amd_gfx1250_TDM.h>)
#define TDM_6ARG 1
#else
#define TDM_6ARG 0
#endif

#if HAVE_TDM
typedef unsigned int v4u __attribute__((ext_vector_type(4)));
typedef int          v4i __attribute__((ext_vector_type(4)));
typedef int          v8i __attribute__((ext_vector_type(8)));

// 2-D tile DMA: global (row-major, 2-byte elements) -> LDS (tile compacted).
// D# per CDNA5 ISA ch.8: group0 = {count=1, lds_addr, global_addr, type=2},
// group1 = {data_size=2B, tensor_dim0/1, tile_dim0/1, tensor_dim0_stride}.
__device__ __forceinline__ void tdm_load_2d(unsigned lds_off, const void* gptr,
                                            unsigned td0, unsigned td1,
                                            unsigned tile0, unsigned tile1,
                                            unsigned stride0) {
  unsigned long long ga = (unsigned long long)(size_t)gptr;
  v4u g0;
  g0[0] = 1u;                                   // count=1, user descriptor
  g0[1] = lds_off;                              // lds_addr (bytes)
  g0[2] = (unsigned)ga;                         // global_addr[31:0]
  g0[3] = (unsigned)((ga >> 32) & 0x01FFFFFFu)  // global_addr[56:32]
        | (2u << 30);                           // type = 2 ("image")

  unsigned long long q0 = (1ull << 16)                               // data_size = 2B
                        | ((unsigned long long)(td0 & 0xFFFFu) << 48);
  unsigned long long q1 = (unsigned long long)(td0 >> 16)
                        | ((unsigned long long)td1 << 16)            // tensor_dim1
                        | ((unsigned long long)(tile0 & 0xFFFFu) << 48);
  unsigned long long q2 = (unsigned long long)(tile1 & 0xFFFFu)      // tile_dim1
                        | ((unsigned long long)stride0 << 32);       // dim0_stride lo32
  unsigned long long q3 = 0;                                         // stride hi / dim1_stride

  v8i g1;
  g1[0] = (int)(unsigned)q0; g1[1] = (int)(q0 >> 32);
  g1[2] = (int)(unsigned)q1; g1[3] = (int)(q1 >> 32);
  g1[4] = (int)(unsigned)q2; g1[5] = (int)(q2 >> 32);
  g1[6] = (int)(unsigned)q3; g1[7] = (int)(q3 >> 32);
  v4i z4 = {0, 0, 0, 0};
#if TDM_6ARG
  v8i z8 = {0, 0, 0, 0, 0, 0, 0, 0};
  __builtin_amdgcn_tensor_load_to_lds(g0, g1, z4, z4, z8, 0);
#else
  __builtin_amdgcn_tensor_load_to_lds(g0, g1, z4, z4, 0);
#endif
}
#endif  // HAVE_TDM

// ---------------------------------------------------------------------------
// WMMA wrapper: D = A(16x32 f16) * B(32x16 f16) + C(16x16 f32)
// ---------------------------------------------------------------------------
__device__ __forceinline__ v8f wmma16(v16h a, v16h b, v8f c) {
  return __builtin_amdgcn_wmma_f32_16x16x32_f16(
      false, a, false, b, (short)0, c, false, false);
}

// A-fragment (16x32, MxK) from row-major f16
__device__ __forceinline__ v16h load_a_f16(const _Float16* __restrict__ A,
                                           int lda, int row0, int k0) {
  int lane = threadIdx.x & 31;
  int m = lane & 15, h = lane >> 4;
  const _Float16* p = A + (size_t)(row0 + m) * lda + k0 + 8 * h;
  v8h lo = *(const v8h*)p;
  v8h hi = *(const v8h*)(p + 16);
  return __builtin_shufflevector(lo, hi, 0,1,2,3,4,5,6,7,8,9,10,11,12,13,14,15);
}

// A-fragment from row-major f32 source (convert in-register)
__device__ __forceinline__ v16h load_a_f32(const float* __restrict__ A,
                                           int lda, int row0, int k0) {
  int lane = threadIdx.x & 31;
  int m = lane & 15, h = lane >> 4;
  const float* p = A + (size_t)(row0 + m) * lda + k0 + 8 * h;
  v16h a;
#pragma unroll
  for (int j = 0; j < 8; ++j) {
    a[j]     = (_Float16)p[j];
    a[8 + j] = (_Float16)p[j + 16];
  }
  return a;
}

// B-fragment (32x16, KxN) from an "n-major" [n][k] source (global or LDS):
// two 8-half (16B) loads per lane: k0+8g..+7 and k0+16+8g..+7
__device__ __forceinline__ v16h load_b_f16(const _Float16* Bn,
                                           int ldb, int n0, int k0) {
  int lane = threadIdx.x & 31;
  int n = lane & 15, g = lane >> 4;
  const _Float16* p = Bn + (size_t)(n0 + n) * ldb + k0 + g * 8;
  v8h lo = *(const v8h*)p;
  v8h hi = *(const v8h*)(p + 16);
  return __builtin_shufflevector(lo, hi, 0,1,2,3,4,5,6,7,8,9,10,11,12,13,14,15);
}

// ---------------------------------------------------------------------------
// Weight transpose + f32->f16 convert:  W[K][N] f32  ->  WT[N][K] f16
// ---------------------------------------------------------------------------
__global__ void cvt_wT(const float* __restrict__ W, _Float16* __restrict__ WT,
                       int K, int N) {
  int idx = blockIdx.x * blockDim.x + threadIdx.x;
  if (idx >= K * N) return;
  int n = idx / K;
  int k = idx - n * K;
  WT[idx] = (_Float16)W[(size_t)k * N + n];
}

// ---------------------------------------------------------------------------
// Projection GEMM: Out_f16[b, M, N] = X_f32[b, M, K] @ WT_f16[N,K]^T
// transpose_out != 0 : store Out transposed as [b, N, M] (for V)
// ---------------------------------------------------------------------------
__global__ void __launch_bounds__(128)
proj_gemm(const float* __restrict__ X, const _Float16* __restrict__ WT,
          _Float16* __restrict__ Out, int transpose_out) {
  const int M = SQ, K = QD, N = IDM;
  int wave = blockIdx.x * 4 + (threadIdx.x >> 5);
  const int tiles_per_b = (M / 16) * (N / 16);
  int b  = wave / tiles_per_b;
  int t  = wave % tiles_per_b;
  int i0 = (t / (N / 16)) * 16;
  int n0 = (t % (N / 16)) * 16;
  int lane = threadIdx.x & 31;
  int nn = lane & 15, hh = lane >> 4;

  const float* Xb = X + (size_t)b * M * K;
  v8f acc = {0.f, 0.f, 0.f, 0.f, 0.f, 0.f, 0.f, 0.f};
#pragma unroll 4
  for (int k0 = 0; k0 < K; k0 += 32) {
    v16h a  = load_a_f32(Xb, K, i0, k0);
    v16h bf = load_b_f16(WT, K, n0, k0);
    acc = wmma16(a, bf, acc);
  }
  if (!transpose_out) {
    _Float16* Ob = Out + (size_t)b * M * N;
#pragma unroll
    for (int r = 0; r < 8; ++r)
      Ob[(size_t)(i0 + hh * 8 + r) * N + n0 + nn] = (_Float16)acc[r];
  } else {  // [b][N][M] : contiguous 8 halves per lane -> one 16B store
    _Float16* Ob = Out + (size_t)b * N * M;
    v8h pk;
#pragma unroll
    for (int r = 0; r < 8; ++r) pk[r] = (_Float16)acc[r];
    *(v8h*)(Ob + (size_t)(n0 + nn) * M + i0 + hh * 8) = pk;
  }
}

// ---------------------------------------------------------------------------
// Flash attention. Block = 4 waves, same batch, adjacent 16-row i-tiles.
// K/V j-chunks (32 wide) staged in LDS by the Tensor Data Mover, double-
// buffered so DMA of chunk n+1 overlaps the 32 WMMAs of chunk n.
// ---------------------------------------------------------------------------
__global__ void __launch_bounds__(128)
flash_attn(const _Float16* __restrict__ Q, const _Float16* __restrict__ Km,
           const _Float16* __restrict__ Vt, const unsigned char* __restrict__ msk,
           _Float16* __restrict__ O) {
  const float scale = 0.0625f;  // 256^-0.5
  const float NEGMAX = -3.4028234e38f;

#if HAVE_TDM
  __shared__ _Float16 ldsK[2][32 * IDM];   // 2 x 16KB : K chunk, [32 j][256 d]
  __shared__ _Float16 ldsV[2][IDM * 32];   // 2 x 16KB : V chunk, [256 d][32 j]
#endif
  __shared__ _Float16 ldsP[4][16 * 32];    // per-wave P bounce (1KB each)

  int widx = threadIdx.x >> 5;
  int b  = blockIdx.x >> 5;                         // 32 blocks per batch
  int i0 = ((blockIdx.x & 31) << 6) + widx * 16;    // 64 rows per block
  int lane = threadIdx.x & 31;
  int nn = lane & 15, hh = lane >> 4;

  const _Float16* Qb = Q  + (size_t)b * SQ * IDM;
  const _Float16* Kb = Km + (size_t)b * SJ * IDM;
  const _Float16* Vb = Vt + (size_t)b * IDM * SJ;
  const unsigned char* mb = msk + (size_t)b * SJ;
  _Float16* pl = &ldsP[widx][0];

  // Q fragments resident for the whole j-loop (8 x 8 VGPRs)
  v16h qa[8];
#pragma unroll
  for (int kc = 0; kc < 8; ++kc) qa[kc] = load_a_f16(Qb, IDM, i0, kc * 32);

  v8f Oacc[16];
  const v8f vz = {0.f, 0.f, 0.f, 0.f, 0.f, 0.f, 0.f, 0.f};
#pragma unroll
  for (int dt = 0; dt < 16; ++dt) Oacc[dt] = vz;
  float mrow[8], lrow[8];
#pragma unroll
  for (int r = 0; r < 8; ++r) { mrow[r] = NEGMAX; lrow[r] = 0.f; }

#if HAVE_TDM
  bool issuer = (widx == 0);
  if (issuer) {  // prologue: chunk 0 into buffer 0
    tdm_load_2d((unsigned)(size_t)&ldsK[0][0], Kb,
                /*td0=*/IDM, /*td1=*/SJ, /*tile0=*/IDM, /*tile1=*/32, /*stride=*/IDM);
    tdm_load_2d((unsigned)(size_t)&ldsV[0][0], Vb,
                /*td0=*/SJ, /*td1=*/IDM, /*tile0=*/32, /*tile1=*/IDM, /*stride=*/SJ);
  }
#endif

  for (int j0 = 0, it = 0; j0 < SJ; j0 += 32, ++it) {
#if HAVE_TDM
    int cur = it & 1, nxt = cur ^ 1;
    if (issuer) __builtin_amdgcn_s_wait_tensorcnt(0);  // chunk `it` resident
    __syncthreads();  // publish buffer `cur`; also proves compute(it-1) done
    if (issuer && (j0 + 32 < SJ)) {                    // overlap: DMA chunk it+1
      tdm_load_2d((unsigned)(size_t)&ldsK[nxt][0], Kb + (size_t)(j0 + 32) * IDM,
                  IDM, SJ, IDM, 32, IDM);
      tdm_load_2d((unsigned)(size_t)&ldsV[nxt][0], Vb + (j0 + 32),
                  SJ, IDM, 32, IDM, SJ);
    }
    const _Float16* kC = &ldsK[cur][0];   // [32][256]
    const _Float16* vC = &ldsV[cur][0];   // [256][32]
#endif

    // S tile 16 x 32 = two 16x16 WMMA accumulators
    v8f s0 = vz, s1 = vz;
#pragma unroll
    for (int kc = 0; kc < 8; ++kc) {
#if HAVE_TDM
      v16h b0 = load_b_f16(kC, IDM, 0,  kc * 32);   // ds reads
      v16h b1 = load_b_f16(kC, IDM, 16, kc * 32);
#else
      v16h b0 = load_b_f16(Kb, IDM, j0,      kc * 32);
      v16h b1 = load_b_f16(Kb, IDM, j0 + 16, kc * 32);
#endif
      s0 = wmma16(qa[kc], b0, s0);
      s1 = wmma16(qa[kc], b1, s1);
    }
    bool v0 = mb[j0 + nn] != 0;
    bool v1 = mb[j0 + 16 + nn] != 0;
#pragma unroll
    for (int r = 0; r < 8; ++r) {
      s0[r] = v0 ? s0[r] * scale : NEGMAX;
      s1[r] = v1 ? s1[r] * scale : NEGMAX;
    }
    // row max across the 16-lane group (row r+8h lives in VGPR r)
    float t[8];
#pragma unroll
    for (int r = 0; r < 8; ++r) t[r] = fmaxf(s0[r], s1[r]);
#pragma unroll
    for (int s = 1; s < 16; s <<= 1)
#pragma unroll
      for (int r = 0; r < 8; ++r) t[r] = fmaxf(t[r], __shfl_xor(t[r], s, 32));

    float sf[8], rs[8];
#pragma unroll
    for (int r = 0; r < 8; ++r) {
      float mnew = fmaxf(mrow[r], t[r]);
      sf[r] = __expf(mrow[r] - mnew);
      mrow[r] = mnew;
      float p0 = __expf(s0[r] - mnew);
      float p1 = __expf(s1[r] - mnew);
      s0[r] = p0; s1[r] = p1;
      rs[r] = p0 + p1;
    }
#pragma unroll
    for (int s = 1; s < 16; s <<= 1)
#pragma unroll
      for (int r = 0; r < 8; ++r) rs[r] += __shfl_xor(rs[r], s, 32);
#pragma unroll
    for (int r = 0; r < 8; ++r) lrow[r] = lrow[r] * sf[r] + rs[r];
#pragma unroll
    for (int dt = 0; dt < 16; ++dt)
#pragma unroll
      for (int r = 0; r < 8; ++r) Oacc[dt][r] *= sf[r];

    // P: C-layout -> A-layout via per-wave LDS bounce
#pragma unroll
    for (int r = 0; r < 8; ++r) {
      pl[(hh * 8 + r) * 32 + nn]      = (_Float16)s0[r];
      pl[(hh * 8 + r) * 32 + 16 + nn] = (_Float16)s1[r];
    }
    v8h plo = *(const v8h*)(pl + nn * 32 + 8 * hh);
    v8h phi = *(const v8h*)(pl + nn * 32 + 16 + 8 * hh);
    v16h pa = __builtin_shufflevector(plo, phi,
                                      0,1,2,3,4,5,6,7,8,9,10,11,12,13,14,15);
    // O += P @ V
#pragma unroll
    for (int dt = 0; dt < 16; ++dt) {
#if HAVE_TDM
      v16h bv = load_b_f16(vC, 32, dt * 16, 0);     // ds reads
#else
      v16h bv = load_b_f16(Vb, SJ, dt * 16, j0);
#endif
      Oacc[dt] = wmma16(pa, bv, Oacc[dt]);
    }
  }

  _Float16* Ob = O + (size_t)b * SQ * IDM;
#pragma unroll
  for (int dt = 0; dt < 16; ++dt) {
#pragma unroll
    for (int r = 0; r < 8; ++r) {
      float val = Oacc[dt][r] / lrow[r];
      Ob[(size_t)(i0 + hh * 8 + r) * IDM + dt * 16 + nn] = (_Float16)val;
    }
  }
}

// ---------------------------------------------------------------------------
// Output projection + residual: Y = O16 @ Wo + bo + X  (f32 out)
// ---------------------------------------------------------------------------
__global__ void __launch_bounds__(128)
out_proj(const _Float16* __restrict__ O16, const _Float16* __restrict__ WoT,
         const float* __restrict__ bo, const float* __restrict__ X,
         float* __restrict__ Y) {
  const int K = IDM, N = QD;
  int wave = blockIdx.x * 4 + (threadIdx.x >> 5);
  int i0 = (wave / (N / 16)) * 16;
  int n0 = (wave % (N / 16)) * 16;
  int lane = threadIdx.x & 31;
  int nn = lane & 15, hh = lane >> 4;

  v8f acc = {0.f, 0.f, 0.f, 0.f, 0.f, 0.f, 0.f, 0.f};
#pragma unroll
  for (int k0 = 0; k0 < K; k0 += 32) {
    v16h a  = load_a_f16(O16, K, i0, k0);
    v16h bf = load_b_f16(WoT, K, n0, k0);
    acc = wmma16(a, bf, acc);
  }
  int col = n0 + nn;
  float bias = bo[col];
#pragma unroll
  for (int r = 0; r < 8; ++r) {
    size_t row = (size_t)(i0 + hh * 8 + r);
    Y[row * N + col] = acc[r] + bias + X[row * N + col];
  }
}

// ---------------------------------------------------------------------------
extern "C" void kernel_launch(void* const* d_in, const int* in_sizes, int n_in,
                              void* d_out, int out_size, void* d_ws, size_t ws_size,
                              hipStream_t stream) {
  const float* x   = (const float*)d_in[0];
  const float* ctx = (const float*)d_in[1];
  const unsigned char* mask = (const unsigned char*)d_in[2];
  const float* Wq = (const float*)d_in[3];
  const float* Wk = (const float*)d_in[4];
  const float* Wv = (const float*)d_in[5];
  const float* Wo = (const float*)d_in[6];
  const float* bo = (const float*)d_in[7];
  float* out = (float*)d_out;

  char* ws = (char*)d_ws;
  _Float16* WqT = (_Float16*)(ws + 0);          // [256][512] f16
  _Float16* WkT = (_Float16*)(ws + 262144);     // [256][512]
  _Float16* WvT = (_Float16*)(ws + 524288);     // [256][512]
  _Float16* WoT = (_Float16*)(ws + 786432);     // [512][256]
  _Float16* q16 = (_Float16*)(ws + 1048576);    // [8][2048][256]
  _Float16* k16 = (_Float16*)(ws + 9437184);    // [8][2048][256]
  _Float16* vT  = (_Float16*)(ws + 17825792);   // [8][256][2048]
  _Float16* o16 = (_Float16*)(ws + 26214400);   // [8][2048][256]

  // 1) transpose/convert weights to f16 [N][K]
  cvt_wT<<<512, 256, 0, stream>>>(Wq, WqT, QD, IDM);
  cvt_wT<<<512, 256, 0, stream>>>(Wk, WkT, QD, IDM);
  cvt_wT<<<512, 256, 0, stream>>>(Wv, WvT, QD, IDM);
  cvt_wT<<<512, 256, 0, stream>>>(Wo, WoT, IDM, QD);

  // 2) projections (16384 wave-tiles each -> 4096 blocks of 4 waves)
  proj_gemm<<<4096, 128, 0, stream>>>(x,   WqT, q16, 0);
  proj_gemm<<<4096, 128, 0, stream>>>(ctx, WkT, k16, 0);
  proj_gemm<<<4096, 128, 0, stream>>>(ctx, WvT, vT,  1);  // stored [d][j]

  // 3) flash attention (8 batches x 32 blocks, 4 waves each)
  flash_attn<<<256, 128, 0, stream>>>(q16, k16, vT, mask, o16);

  // 4) output projection + bias + residual (32768 wave-tiles -> 8192 blocks)
  out_proj<<<8192, 128, 0, stream>>>(o16, WoT, bo, x, out);
}
